// MiniMultiNet_46351287058819
// MI455X (gfx1250) — matrix operand order repaired and checked by reading, not compiled, not run on hardware
//
#include <hip/hip_runtime.h>

typedef __attribute__((ext_vector_type(16))) _Float16 v16h;
typedef __attribute__((ext_vector_type(2)))  _Float16 v2h;
typedef __attribute__((ext_vector_type(8)))  float    v8f;

// ---------------- workspace layout (float offsets) ----------------
// xT : [2][3026][512]   h0 : [16][3026][512]  h1 : [16][785][512]
// h2 : [32][197][512]   h3 : [32][49][512]
#define XT_OFF 0ull
#define H0_OFF 3098624ull            // 2*3026*512
#define H1_OFF 27887616ull           // H0 + 16*3026*512
#define H2_OFF 34318336ull           // H1 + 16*785*512
#define H3_OFF 37545984ull           // H2 + 32*197*512
// total = 38,348,800 floats = ~147 MB of d_ws

// ---- prep: transpose x [512][2][3025] -> xT [2][3026][512], zero pad columns ----
__global__ void __launch_bounds__(256) prep_kernel(const float* __restrict__ x,
                                                   float* __restrict__ ws) {
    __shared__ float tile[32][33];
    const int blk = blockIdx.x;
    if (blk < 1520) {                               // 95 pixel-tiles x 16 batch-tiles
        const int p0 = (blk % 95) * 32;
        const int b0 = (blk / 95) * 32;
        const int tx = threadIdx.x & 31;
        const int ty = threadIdx.x >> 5;            // 0..7
        float* xT = ws + XT_OFF;
        for (int c = 0; c < 2; ++c) {
            __syncthreads();
            for (int i = 0; i < 4; ++i) {
                const int bl = ty + i * 8;
                const int p = p0 + tx;
                tile[bl][tx] = (p < 3025) ? x[((size_t)(b0 + bl) * 2 + c) * 3025 + p] : 0.0f;
            }
            __syncthreads();
            for (int i = 0; i < 4; ++i) {
                const int pl = ty + i * 8;
                const int p = p0 + pl;
                if (p < 3025)
                    xT[((size_t)c * 3026 + p) * 512 + b0 + tx] = tile[tx][pl];
            }
        }
    } else {                                        // 66 pad-zero blocks
        const int bi = blk - 1520;
        for (int t = threadIdx.x; t < 512; t += 256) {
            if (bi < 2)
                ws[XT_OFF + ((size_t)bi * 3026 + 3025) * 512 + t] = 0.0f;
            else if (bi < 18)
                ws[H0_OFF + ((size_t)(bi - 2) * 3026 + 3025) * 512 + t] = 0.0f;
            else if (bi < 34)
                ws[H1_OFF + ((size_t)(bi - 18) * 785 + 784) * 512 + t] = 0.0f;
            else
                ws[H2_OFF + ((size_t)(bi - 34) * 197 + 196) * 512 + t] = 0.0f;
        }
    }
}

// ---------------- fused conv(+bias) -> pool(mean/9) -> BN -> ReLU ----------------
// in  : [CIN ][PinPad ][512] (pad pixel = PinPad-1, zeroed)
// out : [COUT][PoutPad][512]
// GEMM per pooled pixel q: D[b, o] = sum_j sum_ck in[c, idx_conv[k, p_j], b] * W[o, ck]
// WMMA: M=16 batches per wave, N=16 channels per tile, K consumed 32 at a time (f16).
template<int CIN, int COUT, int HASPOOL>
__global__ void __launch_bounds__(256) conv_kernel(
    const float* __restrict__ in, float* __restrict__ out,
    const float* __restrict__ W, const float* __restrict__ bias,
    const float* __restrict__ gamma, const float* __restrict__ beta,
    const float* __restrict__ rmean, const float* __restrict__ rvar,
    const int* __restrict__ idx_conv, const int* __restrict__ idx_pool,
    int PinPad, int Pconv, int Pout, int PoutPad)
{
    constexpr int KC     = CIN * 9;
    constexpr int KPAD   = (KC + 31) & ~31;      // multiple of 32 (WMMA K-depth)
    constexpr int NCHUNK = KPAD / 32;
    constexpr int NTILES = COUT / 16;
    constexpr int NJ     = HASPOOL ? 9 : 1;

    __shared__ v2h   sWp[(KPAD / 2) * COUT];     // weight f16 pairs: (k,k+1) at col n
    __shared__ int   sCol[NJ][KPAD];             // gather column bases per pool-neighbor
    __shared__ float sT[8][16][17];              // per-wave store-transpose tile

    const int tid  = threadIdx.x;
    const int lane = tid & 31;
    const int w    = tid >> 5;                   // wave id 0..7
    const int q    = blockIdx.x;                 // output pixel
    const int PinGrid = PinPad - 1;              // zero slot pixel

    // stage transposed, zero-padded weights as packed f16 pairs: sWp[kp*COUT+n]
    for (int i = tid; i < (KPAD / 2) * COUT; i += 256) {
        const int kp = i / COUT;
        const int n  = i - kp * COUT;
        const int k  = 2 * kp;
        const float w0 = (k < KC) ? W[n * KC + k] : 0.0f;
        const float w1 = (k + 1 < KC) ? W[n * KC + k + 1] : 0.0f;
        v2h p; p.x = (_Float16)w0; p.y = (_Float16)w1;
        sWp[i] = p;
    }

    // gather column tables for every valid pool-neighbor (block-uniform control)
    int nvalid = 0;
    if (HASPOOL) {
        for (int j = 0; j < 9; ++j) {
            const int pj = idx_pool[j * Pout + q];       // uniform across block
            if (pj < Pconv) {                            // skip pool pad slot
                for (int k = tid; k < KPAD; k += 256) {
                    int pix = PinGrid;                   // zero slot
                    int c = 0;
                    if (k < KC) { c = k / 9; const int t = k - c * 9; pix = idx_conv[t * Pconv + pj]; }
                    sCol[nvalid][k] = (c * PinPad + pix) * 512;
                }
                ++nvalid;
            }
        }
    } else {
        for (int k = tid; k < KPAD; k += 256) {
            int pix = PinGrid;
            int c = 0;
            if (k < KC) { c = k / 9; const int t = k - c * 9; pix = idx_conv[t * Pconv + q]; }
            sCol[0][k] = (c * PinPad + pix) * 512;
        }
        nvalid = 1;
    }
    __syncthreads();

    v8f acc0 = {};
    v8f acc1 = {};
    const int bbase = blockIdx.y * 128 + w * 16;  // batch tile base for this wave
    const int ncol  = lane & 15;                  // M row (A) / N col (B,C)
    const int hh    = lane >> 4;
    const float* inb = in + bbase + ncol;

    for (int kc = 0; kc < NCHUNK; ++kc) {
        const int kbase = kc * 32;
        // B fragments: contiguous k order (not the ISA table order); lane-half hh holds K-run [16h,16h+16)
        const int kp0 = (kbase >> 1) + 8 * hh;    // pair index base
        v16h b0, b1;
        #pragma unroll
        for (int v = 0; v < 8; ++v) {
            const v2h p = sWp[(kp0 + v) * COUT + ncol];
            b0[2 * v]     = p.x;
            b0[2 * v + 1] = p.y;
        }
        if (NTILES == 2) {
            #pragma unroll
            for (int v = 0; v < 8; ++v) {
                const v2h p = sWp[(kp0 + v) * COUT + 16 + ncol];
                b1[2 * v]     = p.x;
                b1[2 * v + 1] = p.y;
            }
        }
        for (int jv = 0; jv < nvalid; ++jv) {     // uniform trip count
            // A fragment: contiguous k order (not the ISA table order):
            // lane-half hh holds K-run [16h,16h+16) (matches B)
            v16h a;
            #pragma unroll
            for (int v = 0; v < 8; ++v) {
                const int krel = 16 * hh + 2 * v;
                const int k = kbase + krel;
                a[2 * v]     = (_Float16)inb[sCol[jv][k]];
                a[2 * v + 1] = (_Float16)inb[sCol[jv][k + 1]];
            }
            acc0 = __builtin_amdgcn_wmma_f32_16x16x32_f16(false, a, false, b0,
                                                          (short)0, acc0, false, false);
            if (NTILES == 2) {
                acc1 = __builtin_amdgcn_wmma_f32_16x16x32_f16(false, a, false, b1,
                                                              (short)0, acc1, false, false);
            }
        }
    }

    // epilogue: mean/9 (if pooled) + bias*nvalid/9, BN, ReLU; coalesced store via LDS
    const float inv = HASPOOL ? (1.0f / 9.0f) : 1.0f;
    for (int nt = 0; nt < NTILES; ++nt) {
        const int n = nt * 16 + ncol;
        const float sc = gamma[n] * rsqrtf(rvar[n] + 1e-5f);
        const float sh = beta[n] - rmean[n] * sc;
        const float bterm = bias[n] * (float)nvalid * inv;
        v8f acc;
        if (nt == 0) acc = acc0; else acc = acc1;
        __syncthreads();
        for (int r = 0; r < 8; ++r) {
            float v = acc[r] * inv + bterm;           // conv+bias, pooled
            v = v * sc + sh;                          // BN (eval mode)
            v = v > 0.0f ? v : 0.0f;                  // ReLU
            sT[w][ncol][r + (hh << 3)] = v;           // [n][m]
        }
        __syncthreads();
        for (int r = 0; r < 8; ++r) {
            const int nl = r + (hh << 3);             // channel row 0..15
            const int ch = nt * 16 + nl;
            out[((size_t)ch * PoutPad + q) * 512 + bbase + ncol] = sT[w][nl][ncol];
        }
    }
}

// ---------------- mean over pixels + MLP heads ----------------
__global__ void __launch_bounds__(256) heads_kernel(
    const float* __restrict__ h3,   // [32][49][512]
    const float* __restrict__ We1, const float* __restrict__ be1,
    const float* __restrict__ We2, const float* __restrict__ be2,
    const float* __restrict__ Wf,  const float* __restrict__ bf,
    const float* __restrict__ Wi,  const float* __restrict__ bi,
    const float* __restrict__ Wd,  const float* __restrict__ bd,
    const float* __restrict__ Wc,  const float* __restrict__ bc,
    float* __restrict__ outp)
{
    const int b = blockIdx.x * 256 + threadIdx.x;
    if (b >= 512) return;

    float o[32];
    for (int c = 0; c < 32; ++c) {
        float s = 0.0f;
        for (int q = 0; q < 49; ++q) s += h3[((size_t)c * 49 + q) * 512 + b];
        o[c] = s * (1.0f / 49.0f);
    }
    // energy = relu(o @ We1^T + be1) @ We2^T + be2
    float e = be2[0];
    for (int i = 0; i < 32; ++i) {
        float t = be1[i];
        for (int j = 0; j < 32; ++j) t += We1[i * 32 + j] * o[j];
        t = t > 0.0f ? t : 0.0f;
        e += We2[i] * t;
    }
    outp[b] = e;
    // f = relu(o @ Wf^T + bf)
    float f[64];
    for (int i = 0; i < 64; ++i) {
        float t = bf[i];
        for (int j = 0; j < 32; ++j) t += Wf[i * 32 + j] * o[j];
        f[i] = t > 0.0f ? t : 0.0f;
    }
    for (int i = 0; i < 2; ++i) {
        float ti = bi[i], td = bd[i];
        for (int j = 0; j < 64; ++j) { ti += Wi[i * 64 + j] * f[j]; td += Wd[i * 64 + j] * f[j]; }
        outp[512 + b * 2 + i]        = ti;   // impact
        outp[512 + 1024 + b * 2 + i] = td;   // direction
    }
    // log_softmax(o @ Wc^T + bc)
    float z0 = bc[0], z1 = bc[1];
    for (int j = 0; j < 32; ++j) { z0 += Wc[j] * o[j]; z1 += Wc[32 + j] * o[j]; }
    const float mx  = z0 > z1 ? z0 : z1;
    const float lse = mx + logf(expf(z0 - mx) + expf(z1 - mx));
    outp[512 + 2048 + b * 2 + 0] = z0 - lse;
    outp[512 + 2048 + b * 2 + 1] = z1 - lse;
}

extern "C" void kernel_launch(void* const* d_in, const int* in_sizes, int n_in,
                              void* d_out, int out_size, void* d_ws, size_t ws_size,
                              hipStream_t stream) {
    (void)in_sizes; (void)n_in; (void)out_size; (void)ws_size;
    const float* x = (const float*)d_in[0];
    float* ws = (float*)d_ws;
    float* xT = ws + XT_OFF;
    float* h0 = ws + H0_OFF;
    float* h1 = ws + H1_OFF;
    float* h2 = ws + H2_OFF;
    float* h3 = ws + H3_OFF;

    prep_kernel<<<1586, 256, 0, stream>>>(x, ws);

    // layer 0: conv -> bn -> relu (no pool)
    conv_kernel<2, 16, 0><<<dim3(3025, 4), 256, 0, stream>>>(
        xT, h0,
        (const float*)d_in[1], (const float*)d_in[2], (const float*)d_in[3],
        (const float*)d_in[4], (const float*)d_in[5], (const float*)d_in[6],
        (const int*)d_in[7], (const int*)0,
        3026, 3025, 3025, 3026);
    // layer 1: conv -> pool -> bn -> relu  (55x55 -> 28x28)
    conv_kernel<16, 16, 1><<<dim3(784, 4), 256, 0, stream>>>(
        h0, h1,
        (const float*)d_in[8], (const float*)d_in[9], (const float*)d_in[10],
        (const float*)d_in[11], (const float*)d_in[12], (const float*)d_in[13],
        (const int*)d_in[14], (const int*)d_in[15],
        3026, 3025, 784, 785);
    // layer 2: (28x28 -> 14x14)
    conv_kernel<16, 32, 1><<<dim3(196, 4), 256, 0, stream>>>(
        h1, h2,
        (const float*)d_in[16], (const float*)d_in[17], (const float*)d_in[18],
        (const float*)d_in[19], (const float*)d_in[20], (const float*)d_in[21],
        (const int*)d_in[22], (const int*)d_in[23],
        785, 784, 196, 197);
    // layer 3: (14x14 -> 7x7)
    conv_kernel<32, 32, 1><<<dim3(49, 4), 256, 0, stream>>>(
        h2, h3,
        (const float*)d_in[24], (const float*)d_in[25], (const float*)d_in[26],
        (const float*)d_in[27], (const float*)d_in[28], (const float*)d_in[29],
        (const int*)d_in[30], (const int*)d_in[31],
        197, 196, 49, 49);

    heads_kernel<<<2, 256, 0, stream>>>(
        h3,
        (const float*)d_in[32], (const float*)d_in[33], (const float*)d_in[34],
        (const float*)d_in[35], (const float*)d_in[36], (const float*)d_in[37],
        (const float*)d_in[38], (const float*)d_in[39], (const float*)d_in[40],
        (const float*)d_in[41], (const float*)d_in[42], (const float*)d_in[43],
        (float*)d_out);
}
